// VQHeadEMA_44590350467541
// MI455X (gfx1250) — matrix operand order repaired and checked by reading, not compiled, not run on hardware
//
#include <hip/hip_runtime.h>
#include <hip/hip_bf16.h>

typedef __attribute__((ext_vector_type(16))) __bf16 v16bf;
typedef __attribute__((ext_vector_type(8)))  float  v8f;

#define DIM       256
#define KC        1024
#define NROWS     65536          // 16 * 4096
#define NTILES    (NROWS / 16)   // 4096
#define DECAYF    0.99f
#define ONEMDF    0.01f
#define EPSF      1e-5f

#define CHUNK_NT    4                      // nt-groups staged per LDS chunk
#define NCHUNK      (64 / CHUNK_NT)        // 16 chunks
#define CHUNK_ELEMS (CHUNK_NT * 8 * 32 * 16)   // 16384 bf16 = 32 KB
#define CHUNK_BYTES (CHUNK_ELEMS * 2)

// ---------------------------------------------------------------------------
// Kernel 1: codebook prep.
//  - csq[k] = ||codebook[k]||^2
//  - Bws    = bf16 codebook pre-swizzled into a contiguous-k WMMA B-matrix
//             (32x16 KxN) per-lane layout:
//             layout index = ((ntile*8 + s)*32 + lane)*16 + j
//             where lane = half*16 + (k&15), d = s*32 + half*16 + j.
//  - Seed EMA accumulators directly in d_out:
//             out_cs = 0.99*cluster_size, out_ea = 0.99*embed_avg
// ---------------------------------------------------------------------------
__global__ __launch_bounds__(256) void vq_prep(
    const float* __restrict__ codebook,
    const float* __restrict__ cluster_size,
    const float* __restrict__ embed_avg,
    __bf16* __restrict__ Bws,
    float* __restrict__ csq,
    float* __restrict__ out_cs,
    float* __restrict__ out_ea)
{
    __shared__ float red[256];
    const int k = blockIdx.x;     // codebook row
    const int d = threadIdx.x;    // dim

    const float c = codebook[k * DIM + d];

    // ||c||^2 block reduction
    red[d] = c * c;
    __syncthreads();
    #pragma unroll
    for (int off = 128; off > 0; off >>= 1) {
        if (d < off) red[d] += red[d + off];
        __syncthreads();
    }
    if (d == 0) csq[k] = red[0];

    // bf16 swizzled store for WMMA B operand
    const int nt  = k >> 4;
    const int nlo = k & 15;
    const int s   = d >> 5;
    const int r   = d & 31;
    const int hf  = r >> 4;
    const int j   = r & 15;
    Bws[(((nt * 8 + s) * 32) + (hf * 16 + nlo)) * 16 + j] = (__bf16)c;

    // EMA seeds (accumulated by atomics in vq_assign)
    out_ea[k * DIM + d] = DECAYF * embed_avg[k * DIM + d];
    if (d == 0) out_cs[k] = DECAYF * cluster_size[k];
}

// ---------------------------------------------------------------------------
// Kernel 2: assignment via bf16 WMMA + gather + EMA scatter.
// One wave (32 lanes) per 16-row z tile; 8 waves per block.
// The bf16 codebook is staged block-cooperatively into LDS in 32 KB chunks
// using async global->LDS copies (ASYNCcnt), so all 8 waves share one global
// read of each codebook chunk and WMMA B operands come from ds_load_b128.
// ---------------------------------------------------------------------------
__global__ __launch_bounds__(256) void vq_assign(
    const float* __restrict__ z,
    const float* __restrict__ codebook,
    const __bf16* __restrict__ Bws,
    const float* __restrict__ csq,
    float* __restrict__ out_zq,
    float* __restrict__ out_idx,
    float* __restrict__ out_cs,
    float* __restrict__ out_ea)
{
    __shared__ __bf16 lds_b[CHUNK_ELEMS];   // 32 KB staged codebook chunk
    __shared__ int    lds_idx[8][16];

    const int tid  = threadIdx.x;
    const int wave = tid >> 5;
    const int lane = tid & 31;
    const int half = lane >> 4;
    const int mlo  = lane & 15;

    const int  tile     = blockIdx.x * 8 + wave;
    const long row_base = (long)tile * 16;

    // -------- Load A tiles: 16x32 bf16 per tile, 8 tiles cover K-dim 256 ----
    // A order used here (contiguous, same as Bws): lanes 0-15 hold K = [0..16), lanes 16-31 hold
    // K = [16..32) of each 32-wide slab (row M = lane & 15).
    const float* zrow = z + (row_base + mlo) * DIM;
    v16bf a[8];
    #pragma unroll
    for (int s = 0; s < 8; ++s) {
        const float* p0 = zrow + s * 32 + half * 16;        // first 8 K values
        const float* p1 = zrow + s * 32 + half * 16 + 8;   // second 8 K values
        float4 q0 = ((const float4*)p0)[0];
        float4 q1 = ((const float4*)p0)[1];
        float4 q2 = ((const float4*)p1)[0];
        float4 q3 = ((const float4*)p1)[1];
        v16bf t;
        t[0]=(__bf16)q0.x;  t[1]=(__bf16)q0.y;  t[2]=(__bf16)q0.z;  t[3]=(__bf16)q0.w;
        t[4]=(__bf16)q1.x;  t[5]=(__bf16)q1.y;  t[6]=(__bf16)q1.z;  t[7]=(__bf16)q1.w;
        t[8]=(__bf16)q2.x;  t[9]=(__bf16)q2.y;  t[10]=(__bf16)q2.z; t[11]=(__bf16)q2.w;
        t[12]=(__bf16)q3.x; t[13]=(__bf16)q3.y; t[14]=(__bf16)q3.z; t[15]=(__bf16)q3.w;
        a[s] = t;
    }

    // -------- Scan all 64 codebook column tiles, track argmax score --------
    // score[m][n] = z_m . c_n - 0.5*||c_n||^2  (argmax == argmin distance)
    float best[8];
    int   bidx[8];
    #pragma unroll
    for (int v = 0; v < 8; ++v) { best[v] = -3.4e38f; bidx[v] = 0; }

    // 32-bit LDS offset of the staging buffer (low 32 bits of generic ptr)
    const unsigned lds_base = (unsigned)(uintptr_t)(&lds_b[0]);

    for (int chunk = 0; chunk < NCHUNK; ++chunk) {
        __syncthreads();   // all waves done reading the previous chunk

        // ---- async stage: Bws[chunk*16384 .. +16384) bf16 -> LDS ----------
        // 256 threads x 16 B x 8 rounds = 32 KB, tracked by ASYNCcnt.
        {
            const unsigned gbase = (unsigned)chunk * (unsigned)CHUNK_BYTES;
            #pragma unroll
            for (int r = 0; r < 8; ++r) {
                const unsigned boff = (unsigned)(r * 4096 + tid * 16);
                const unsigned loff = lds_base + boff;
                const unsigned goff = gbase + boff;
                asm volatile("global_load_async_to_lds_b128 %0, %1, %2"
                             :: "v"(loff), "v"(goff), "s"(Bws)
                             : "memory");
            }
            asm volatile("s_wait_asynccnt 0x0" ::: "memory");
        }
        __syncthreads();

        // ---- consume: 4 nt groups x 8 WMMA each, B from LDS ---------------
        #pragma unroll
        for (int lnt = 0; lnt < CHUNK_NT; ++lnt) {
            const int   nt = chunk * CHUNK_NT + lnt;
            const int   n  = nt * 16 + mlo;     // this lane's codebook column
            const float c2 = csq[n];

            v8f acc;
            #pragma unroll
            for (int v = 0; v < 8; ++v) acc[v] = -0.5f * c2;

            const __bf16* lb = lds_b + ((lnt * 8) * 32 + lane) * 16;
            v16bf bcur = *(const v16bf*)lb;
            #pragma unroll
            for (int s = 0; s < 8; ++s) {
                v16bf bnext = (s < 7) ? *(const v16bf*)(lb + (s + 1) * 512)
                                      : bcur;
                acc = __builtin_amdgcn_wmma_f32_16x16x32_bf16(
                    false, a[s], false, bcur, (short)0, acc, false, false);
                bcur = bnext;
            }

            #pragma unroll
            for (int v = 0; v < 8; ++v) {
                if (acc[v] > best[v]) { best[v] = acc[v]; bidx[v] = n; }
            }
        }
    }

    // -------- Cross-lane argmax reduction within each 16-lane half ---------
    // Lane half h holds rows M = v + h*8; columns are spread over 16 lanes.
    #pragma unroll
    for (int off = 8; off >= 1; off >>= 1) {
        #pragma unroll
        for (int v = 0; v < 8; ++v) {
            float os = __shfl_xor(best[v], off);
            int   oi = __shfl_xor(bidx[v], off);
            if (os > best[v] || (os == best[v] && oi < bidx[v])) {
                best[v] = os; bidx[v] = oi;
            }
        }
    }

    // -------- Emit indices, counts; stash idx in LDS for gather phase ------
    if (mlo == 0) {
        #pragma unroll
        for (int v = 0; v < 8; ++v) {
            const int m = half * 8 + v;
            lds_idx[wave][m] = bidx[v];
            out_idx[row_base + m] = (float)bidx[v];
            atomicAdd(out_cs + bidx[v], ONEMDF);
        }
    }
    __syncthreads();

    // -------- Gather z_q = codebook[idx]; scatter 0.01*z into embed_avg ----
    const int d0 = lane * 8;  // 32 lanes x 8 floats = 256 dims
    for (int r = 0; r < 16; ++r) {
        const int   idx = lds_idx[wave][r];
        const long  row = row_base + r;
        const float* cr = codebook + (long)idx * DIM;
        const float* zr = z + row * DIM;
        float*       oq = out_zq + row * DIM;
        float*       ea = out_ea + (long)idx * DIM;

        float4 c0 = ((const float4*)(cr + d0))[0];
        float4 c1 = ((const float4*)(cr + d0))[1];
        ((float4*)(oq + d0))[0] = c0;
        ((float4*)(oq + d0))[1] = c1;

        float4 z0 = ((const float4*)(zr + d0))[0];
        float4 z1 = ((const float4*)(zr + d0))[1];
        float zv[8] = {z0.x, z0.y, z0.z, z0.w, z1.x, z1.y, z1.z, z1.w};
        #pragma unroll
        for (int j = 0; j < 8; ++j)
            atomicAdd(ea + d0 + j, ONEMDF * zv[j]);
    }
}

// ---------------------------------------------------------------------------
// Kernel 3: new_codebook = new_embed_avg / (new_cluster_size + eps)
// ---------------------------------------------------------------------------
__global__ __launch_bounds__(256) void vq_final(
    const float* __restrict__ out_cs,
    const float* __restrict__ out_ea,
    float* __restrict__ out_cb)
{
    const int k = blockIdx.x;
    const int d = threadIdx.x;
    out_cb[k * DIM + d] = out_ea[k * DIM + d] / (out_cs[k] + EPSF);
}

// ---------------------------------------------------------------------------
extern "C" void kernel_launch(void* const* d_in, const int* in_sizes, int n_in,
                              void* d_out, int out_size, void* d_ws, size_t ws_size,
                              hipStream_t stream) {
    const float* z            = (const float*)d_in[0];
    const float* codebook     = (const float*)d_in[1];
    const float* cluster_size = (const float*)d_in[2];
    const float* embed_avg    = (const float*)d_in[3];

    // Output tuple, concatenated flat (all float32):
    float* out      = (float*)d_out;
    float* out_zq   = out;                         // 16*4096*256 = 16777216
    float* out_idx  = out + 16777216;              // 16*4096     =    65536
    float* out_cb   = out + 16777216 + 65536;      // 1024*256    =   262144
    float* out_cs   = out_cb + 262144;             // 1024
    float* out_ea   = out_cs + 1024;               // 1024*256    =   262144

    // Workspace: swizzled bf16 codebook (512 KB) + csq (4 KB)
    __bf16* Bws = (__bf16*)d_ws;
    float*  csq = (float*)((char*)d_ws + (size_t)KC * DIM * sizeof(__bf16));

    vq_prep  <<<KC,          DIM, 0, stream>>>(codebook, cluster_size, embed_avg,
                                               Bws, csq, out_cs, out_ea);
    vq_assign<<<NTILES / 8,  256, 0, stream>>>(z, codebook, Bws, csq,
                                               out_zq, out_idx, out_cs, out_ea);
    vq_final <<<KC,          DIM, 0, stream>>>(out_cs, out_ea, out_cb);
}
